// SelfAttention_43550968382290
// MI455X (gfx1250) — matrix operand order repaired and checked by reading, not compiled, not run on hardware
//
#include <hip/hip_runtime.h>

typedef __attribute__((ext_vector_type(16))) _Float16 v16h;
typedef __attribute__((ext_vector_type(8)))  _Float16 v8h;
typedef __attribute__((ext_vector_type(8)))  float    v8f;

#define WMMA16(A, B, C) \
    __builtin_amdgcn_wmma_f32_16x16x32_f16(false, (A), false, (B), (short)0, (C), false, false)

#define BATCH 4
#define CH 64
#define NVOX 4096
#define NGROUP 32

// ---------------------------------------------------------------------------
// Kernel 1: GroupNorm statistics. One block per (batch, group); each group is
// 2 contiguous channels x 4096 voxels = 8192 contiguous floats.
// ---------------------------------------------------------------------------
__global__ __launch_bounds__(256) void gn_stats_kernel(
    const float* __restrict__ x, float* __restrict__ mean, float* __restrict__ rstd) {
  int bg = blockIdx.x;                       // 0..127
  const float* base = x + (size_t)bg * 8192; // (b*64 + g*2)*4096
  float s = 0.f, ss = 0.f;
  for (int i = threadIdx.x; i < 8192; i += 256) {
    float v = base[i];
    s += v; ss += v * v;
  }
  __shared__ float rs[256], rss[256];
  rs[threadIdx.x] = s; rss[threadIdx.x] = ss;
  __syncthreads();
  for (int off = 128; off > 0; off >>= 1) {
    if (threadIdx.x < off) {
      rs[threadIdx.x]  += rs[threadIdx.x + off];
      rss[threadIdx.x] += rss[threadIdx.x + off];
    }
    __syncthreads();
  }
  if (threadIdx.x == 0) {
    float mu  = rs[0] * (1.f / 8192.f);
    float var = rss[0] * (1.f / 8192.f) - mu * mu;
    mean[bg] = mu;
    rstd[bg] = rsqrtf(var + 1e-5f);
  }
}

// ---------------------------------------------------------------------------
// Kernel 2: normalize + QKV projection (1x1x1 conv as GEMM), write f16:
//   qh[b][n][c] = (0.125 * log2e) * q   (1/sqrt(C) and exp->exp2 folded in)
//   kh[b][n][c] = k
//   vh[b][c][n] = v                     (channel-major for V A-fragments)
// ---------------------------------------------------------------------------
__global__ __launch_bounds__(256) void qkv_kernel(
    const float* __restrict__ x, const float* __restrict__ gnw,
    const float* __restrict__ gnb, const float* __restrict__ wqkv,
    const float* __restrict__ mean, const float* __restrict__ rstd,
    _Float16* __restrict__ qh, _Float16* __restrict__ kh, _Float16* __restrict__ vh) {
  __shared__ float nrm[64][65];
  int blk = blockIdx.x;
  int b  = blk >> 6;
  int n0 = (blk & 63) << 6;
  int t  = threadIdx.x;

  {
    int c  = t >> 2;
    int v0 = (t & 3) << 4;
    int bg = b * NGROUP + (c >> 1);
    float mu = mean[bg], rsd = rstd[bg];
    float w = gnw[c], bb = gnb[c];
    const float* xp = x + ((size_t)b * CH + c) * NVOX + n0 + v0;
#pragma unroll
    for (int i = 0; i < 16; ++i)
      nrm[c][v0 + i] = (xp[i] - mu) * rsd * w + bb;
  }
  __syncthreads();

  int v  = t & 63;
  int dg = t >> 6;
  int n  = n0 + v;
  const float QSCALE = 0.125f * 1.44269504088896f; // (1/sqrt(C)) * log2(e)
  for (int j = 0; j < 48; ++j) {
    int d = dg * 48 + j;
    const float* wr = wqkv + d * CH;
    float acc = 0.f;
#pragma unroll 8
    for (int c = 0; c < CH; ++c) acc += wr[c] * nrm[c][v];
    if (d < 64) {
      qh[((size_t)b * NVOX + n) * CH + d] = (_Float16)(QSCALE * acc);
    } else if (d < 128) {
      kh[((size_t)b * NVOX + n) * CH + (d - 64)] = (_Float16)acc;
    } else {
      vh[((size_t)b * CH + (d - 128)) * NVOX + n] = (_Float16)acc;
    }
  }
}

// xor-16 half-wave exchange as a pure VALU op (v_permlanex16_b32 with
// identity selects) -- avoids the ds_bpermute/LDS path in the hot loop.
__device__ inline float xor16(float v) {
  unsigned u = __float_as_uint(v);
  u = __builtin_amdgcn_permlanex16(u, u, 0x76543210u, 0xfedcba98u, false, false);
  return __uint_as_float(u);
}

// raw v_exp_f32 (2^x); softmax runs in the log2 domain
__device__ inline float exp2raw(float v) { return __builtin_amdgcn_exp2f(v); }

// load a 16-bit A-fragment (16x32): halves 0..7 at p[0..7], 8..15 at p[8..15]
__device__ inline v16h load_a16(const _Float16* p) {
  v8h lo = *(const v8h*)p;
  v8h hh = *(const v8h*)(p + 8);
  v16h a;
#pragma unroll
  for (int i = 0; i < 8; ++i) { a[i] = lo[i]; a[i + 8] = hh[i]; }
  return a;
}

__device__ inline void load_kfrags(v16h kf[4], const _Float16* kbase,
                                   int kb, int col, int hi) {
  const _Float16* kr0 = kbase + (size_t)(kb + col) * CH + hi * 16;
  const _Float16* kr1 = kbase + (size_t)(kb + 16 + col) * CH + hi * 16;
  kf[0] = load_a16(kr0);
  kf[1] = load_a16(kr0 + 32);
  kf[2] = load_a16(kr1);
  kf[3] = load_a16(kr1 + 32);
}

__device__ inline void load_vfrags(v16h vf[4], const _Float16* vbase,
                                   int kb, int col, int hi) {
#pragma unroll
  for (int ct = 0; ct < 4; ++ct) {
    const _Float16* vr = vbase + (size_t)(ct * 16 + col) * NVOX + kb + hi * 16;
    vf[ct] = load_a16(vr);
  }
}

// One 32-key chunk: 4 S-WMMAs, online softmax (log2 domain), P^T register
// transpose, 4 O-WMMAs.
#define PROCESS_CHUNK(KF, VF)                                                  \
  {                                                                            \
    v8f f0 = {}, f1 = {};                                                      \
    f0 = WMMA16(KF[0], bq0, f0);                                               \
    f0 = WMMA16(KF[1], bq1, f0);                                               \
    f1 = WMMA16(KF[2], bq0, f1);                                               \
    f1 = WMMA16(KF[3], bq1, f1);                                               \
    float cmax = -1e30f;                                                       \
    _Pragma("unroll") for (int r = 0; r < 8; ++r) {                            \
      cmax = fmaxf(cmax, f0[r]);                                               \
      cmax = fmaxf(cmax, f1[r]);                                               \
    }                                                                          \
    cmax = fmaxf(cmax, xor16(cmax));                                           \
    float mnew  = fmaxf(m_run, cmax);                                          \
    float alpha = exp2raw(m_run - mnew);                                       \
    float p0[8], p1[8];                                                        \
    float ls = 0.f;                                                            \
    _Pragma("unroll") for (int r = 0; r < 8; ++r) {                            \
      p0[r] = exp2raw(f0[r] - mnew); ls += p0[r];                              \
      p1[r] = exp2raw(f1[r] - mnew); ls += p1[r];                              \
    }                                                                          \
    ls += xor16(ls);                                                           \
    l_run = l_run * alpha + ls;                                                \
    m_run = mnew;                                                              \
    _Pragma("unroll") for (int r = 0; r < 8; ++r) {                            \
      O0[r] *= alpha; O1[r] *= alpha; O2[r] *= alpha; O3[r] *= alpha;          \
    }                                                                          \
    v16h pb;                                                                   \
    _Pragma("unroll") for (int i = 0; i < 8; ++i) {                            \
      float send = hi ? p0[i] : p1[i];                                         \
      float recv = xor16(send);                                                \
      pb[i]     = (_Float16)(hi ? recv : p0[i]);                               \
      pb[i + 8] = (_Float16)(hi ? p1[i] : recv);                               \
    }                                                                          \
    O0 = WMMA16(VF[0], pb, O0);                                                \
    O1 = WMMA16(VF[1], pb, O1);                                                \
    O2 = WMMA16(VF[2], pb, O2);                                                \
    O3 = WMMA16(VF[3], pb, O3);                                                \
  }

// ---------------------------------------------------------------------------
// Kernel 3: fused flash attention + output projection + bias + residual.
// 1 wave = 16-query tile; K fragments register-double-buffered across chunks.
// ---------------------------------------------------------------------------
__global__ __launch_bounds__(256) void attn_kernel(
    const float* __restrict__ x, const float* __restrict__ wout,
    const float* __restrict__ bout,
    const _Float16* __restrict__ qh, const _Float16* __restrict__ kh,
    const _Float16* __restrict__ vh, float* __restrict__ out) {
  int wave = (blockIdx.x << 3) + (threadIdx.x >> 5); // 0..1023
  int b  = wave >> 8;
  int qb = (wave & 255) << 4;
  int lane = threadIdx.x & 31;
  int col  = lane & 15;
  int hi   = lane >> 4;

  // Q B-fragments (constant over the key loop)
  const _Float16* qrow = qh + ((size_t)b * NVOX + qb + col) * CH + hi * 16;
  v16h bq0 = *(const v16h*)(qrow);
  v16h bq1 = *(const v16h*)(qrow + 32);

  const _Float16* kbase = kh + (size_t)b * NVOX * CH;
  const _Float16* vbase = vh + (size_t)b * CH * NVOX;

  v8f O0 = {}, O1 = {}, O2 = {}, O3 = {};
  float m_run = -1e30f, l_run = 0.f;

  v16h kA[4], kB[4], vf[4];
  load_kfrags(kA, kbase, 0, col, hi);

  for (int kb = 0; kb < NVOX; kb += 64) {
    // chunk kb: V loads issued first (covered by S-wmma + softmax),
    // next chunk's K loads issued before any compute.
    load_vfrags(vf, vbase, kb, col, hi);
    load_kfrags(kB, kbase, kb + 32, col, hi);
    PROCESS_CHUNK(kA, vf);

    // chunk kb+32
    load_vfrags(vf, vbase, kb + 32, col, hi);
    if (kb + 64 < NVOX) load_kfrags(kA, kbase, kb + 64, col, hi);
    PROCESS_CHUNK(kB, vf);
  }

  float inv = 1.f / l_run;
#pragma unroll
  for (int r = 0; r < 8; ++r) {
    O0[r] *= inv; O1[r] *= inv; O2[r] *= inv; O3[r] *= inv;
  }

  // O (C-layout) -> B-fragments for the projection (same xor-16 transform)
  v16h ob0, ob1;
#pragma unroll
  for (int i = 0; i < 8; ++i) {
    float send = hi ? O0[i] : O1[i];
    float recv = xor16(send);
    ob0[i]     = (_Float16)(hi ? recv : O0[i]);
    ob0[i + 8] = (_Float16)(hi ? O1[i] : recv);
    float send2 = hi ? O2[i] : O3[i];
    float recv2 = xor16(send2);
    ob1[i]     = (_Float16)(hi ? recv2 : O2[i]);
    ob1[i + 8] = (_Float16)(hi ? O3[i] : recv2);
  }

  // Y = Wout * O
  v8f Y[4];
#pragma unroll
  for (int dt = 0; dt < 4; ++dt) {
    v8f y = {};
#pragma unroll
    for (int chunk = 0; chunk < 2; ++chunk) {
      int d = dt * 16 + col;
      const float* wr = wout + d * CH + chunk * 32 + hi * 16;
      v16h aW;
#pragma unroll
      for (int i = 0; i < 8; ++i) {
        aW[i]     = (_Float16)wr[i];
        aW[i + 8] = (_Float16)wr[8 + i];
      }
      y = WMMA16(aW, chunk ? ob1 : ob0, y);
    }
    Y[dt] = y;
  }

  // epilogue: bias + residual
  int n = qb + col;
#pragma unroll
  for (int dt = 0; dt < 4; ++dt) {
#pragma unroll
    for (int r = 0; r < 8; ++r) {
      int d = dt * 16 + r + hi * 8;
      size_t idx = ((size_t)b * CH + d) * NVOX + n;
      out[idx] = Y[dt][r] + bout[d] + x[idx];
    }
  }
}

extern "C" void kernel_launch(void* const* d_in, const int* in_sizes, int n_in,
                              void* d_out, int out_size, void* d_ws, size_t ws_size,
                              hipStream_t stream) {
  const float* x    = (const float*)d_in[0];
  const float* gnw  = (const float*)d_in[1];
  const float* gnb  = (const float*)d_in[2];
  const float* wqkv = (const float*)d_in[3];
  const float* wout = (const float*)d_in[4];
  const float* bout = (const float*)d_in[5];
  float* out = (float*)d_out;

  float* mean = (float*)d_ws;
  float* rstd = mean + 128;
  _Float16* qh = (_Float16*)((char*)d_ws + 1024);
  _Float16* kh = qh + (size_t)BATCH * NVOX * CH;
  _Float16* vh = kh + (size_t)BATCH * NVOX * CH;

  gn_stats_kernel<<<dim3(BATCH * NGROUP), dim3(256), 0, stream>>>(x, mean, rstd);
  qkv_kernel<<<dim3(BATCH * (NVOX / 64)), dim3(256), 0, stream>>>(
      x, gnw, gnb, wqkv, mean, rstd, qh, kh, vh);
  attn_kernel<<<dim3((BATCH * (NVOX / 16)) / 8), dim3(256), 0, stream>>>(
      x, wout, bout, qh, kh, vh, out);
}
